// AttentionLearnFusionDirectModule_85023172592505
// MI455X (gfx1250) — compile-verified, matrix operand order checked by reading, not run on hardware
//
#include <hip/hip_runtime.h>
#include <hip/hip_bf16.h>
#include <math.h>

#define NMEM 8
#define NSEQ 16
#define CCH 256
#define WF 24
#define LF (WF*WF)     /* 576  */
#define WL 96
#define LPIX (WL*WL)   /* 9216 */
#define BN_EPS 1e-5f

typedef __attribute__((ext_vector_type(16))) _Float16 v16h;
typedef __attribute__((ext_vector_type(8)))  _Float16 h8;
typedef __attribute__((ext_vector_type(8)))  float    v8f;

// ---------------------------------------------------------------------------
// Kernel 1: inverse L2 norms along channel dim for test/train features.
// One thread per spatial column; adjacent threads -> adjacent addresses.
// ---------------------------------------------------------------------------
__global__ __launch_bounds__(256) void knorm(const float* __restrict__ te,
                                             const float* __restrict__ tr,
                                             float* __restrict__ inv_te,
                                             float* __restrict__ inv_tr) {
  const int nte = NSEQ * LF;          // 9216 columns of test_feat
  const int ntr = NMEM * NSEQ * LF;   // 73728 columns of train_feats
  int id = blockIdx.x * blockDim.x + threadIdx.x;
  if (id < nte) {
    const float* p = te + (id / LF) * (CCH * LF) + (id % LF);
    float s = 0.f;
    for (int c = 0; c < CCH; ++c) { float x = p[c * LF]; s += x * x; }
    inv_te[id] = 1.0f / sqrtf(s);
  } else if (id < nte + ntr) {
    int j = id - nte;
    const float* p = tr + (j / LF) * (CCH * LF) + (j % LF);
    float s = 0.f;
    for (int c = 0; c < CCH; ++c) { float x = p[c * LF]; s += x * x; }
    inv_tr[j] = 1.0f / sqrtf(s);
  }
}

// ---------------------------------------------------------------------------
// Kernel 2: antialiased bilinear downsample 96x96 -> 24x24 (jax.image.resize
// default antialias=True: triangle kernel of radius 4, edge-renormalized).
// ---------------------------------------------------------------------------
__global__ __launch_bounds__(256) void kldown(const float* __restrict__ lab,
                                              float* __restrict__ ldown) {
  int id = blockIdx.x * blockDim.x + threadIdx.x;
  if (id >= NMEM * NSEQ * LF) return;
  int loc = id % LF;
  int mn  = id / LF;
  int ow = loc / WF, oh = loc % WF;
  const float* src = lab + mn * LPIX;
  float acc = 0.f, norm = 0.f;
  #pragma unroll
  for (int uy = 0; uy < 8; ++uy) {
    int y = 4 * ow - 2 + uy;
    if (y < 0 || y >= WL) continue;
    float wy = 4.0f - fabsf((float)uy - 3.5f);
    #pragma unroll
    for (int ux = 0; ux < 8; ++ux) {
      int x = 4 * oh - 2 + ux;
      if (x < 0 || x >= WL) continue;
      float wx = 4.0f - fabsf((float)ux - 3.5f);
      acc  += wy * wx * src[y * WL + x];
      norm += wy * wx;
    }
  }
  ldown[id] = acc / norm;
}

// ---------------------------------------------------------------------------
// Kernel 3: fused cosine-sim GEMM (WMMA f16 -> f32) + temp-softmax + label
// weighted sum. One workgroup = one (i-tile of 16 test locations, m, n).
// 4 waves; each wave owns 9 of the 36 N-tiles (full 576 train locations).
// Normalization (inv_te[i]*inv_tr[j]) is folded into the sim epilogue, so
// LDS staging is a pure f32->f16 convert. B chunks are double-buffered and
// B fragments are software-pipelined one tile ahead of the WMMA.
// ---------------------------------------------------------------------------
__global__ __launch_bounds__(128) void kattn(const float* __restrict__ te,
                                             const float* __restrict__ tr,
                                             const float* __restrict__ inv_te,
                                             const float* __restrict__ inv_tr,
                                             const float* __restrict__ ldown,
                                             const float* __restrict__ temp_p,
                                             float* __restrict__ pmt_down) {
  // LDS: A panel 16x256 f16 (8 KB) | 2x B chunk 576x32 f16 (2x36 KB) | red.
  // sim tile 16x576 f32 (36 KB) aliases B buffer 0.
  __shared__ __align__(16) unsigned char smem[8192 + 2 * 36864 + 1088];
  _Float16* a_lds = (_Float16*)smem;                  // [row][k], k-contiguous
  _Float16* b0    = (_Float16*)(smem + 8192);         // [j][kk], kk-contiguous
  _Float16* b1    = (_Float16*)(smem + 8192 + 36864);
  float*    sim_lds = (float*)(smem + 8192);          // aliases b0
  float*    redA = (float*)(smem + 8192 + 2 * 36864); // [128]
  float*    redB = redA + 128;                        // [128]

  const int it  = blockIdx.x;           // i tile: 0..35
  const int m   = blockIdx.y;           // memory: 0..7
  const int n   = blockIdx.z;           // seq:    0..15
  const int tid = threadIdx.x;
  const int i0  = it * 16;
  const float temp = temp_p[0];

  const float* teB   = te + n * (CCH * LF);
  const float* trB   = tr + (m * NSEQ + n) * (CCH * LF);
  const float* invTe = inv_te + n * LF;
  const float* invTr = inv_tr + (m * NSEQ + n) * LF;

  // Stage A panel (pure f32->f16): a_lds[row*256 + k] = te[k][i0+row]
  for (int idx = tid; idx < 16 * CCH; idx += 128) {
    int row = idx & 15, k = idx >> 4;
    a_lds[row * CCH + k] = (_Float16)teB[k * LF + i0 + row];
  }

  // B chunk stager: b[j*32 + kk] = tr[kc+kk][j]  (K-minor for b128 fragments)
  auto stageB = [&](_Float16* dst, int kc) {
    for (int idx = tid; idx < 576 * 8; idx += 128) {
      int j  = idx % 576;
      int kg = idx / 576;                       // group of 4 kk
      const float* s = trB + (kc + kg * 4) * LF + j;
      _Float16* d = dst + j * 32 + kg * 4;
      d[0] = (_Float16)s[0];
      d[1] = (_Float16)s[LF];
      d[2] = (_Float16)s[2 * LF];
      d[3] = (_Float16)s[3 * LF];
    }
  };
  stageB(b0, 0);

  v8f acc[9] = {};
  const int lane = tid & 31;
  const int wave = tid >> 5;
  const int g    = lane >> 4;   // K-half select per ISA fragment layout
  const int sub  = lane & 15;   // row (A) / col (B)

  int buf = 0;
  for (int kc = 0; kc < CCH; kc += 32, buf ^= 1) {
    __syncthreads();   // staged chunk visible; previous compute finished
    if (kc + 32 < CCH) stageB(buf ? b0 : b1, kc + 32);
    const _Float16* bcur = buf ? b1 : b0;

    // A fragment: lane holds row=sub, K = kc + {8g..8g+7, 16+8g..16+8g+7}
    const _Float16* ap = a_lds + sub * CCH + kc + 8 * g;
    h8 alo = *(const h8*)ap;
    h8 ahi = *(const h8*)(ap + 16);
    v16h af = __builtin_shufflevector(alo, ahi, 0,1,2,3,4,5,6,7,8,9,10,11,12,13,14,15);

    // Software-pipelined B fragments: load tile q+1 before WMMA on tile q.
    const _Float16* bp0 = bcur + (wave * 9 * 16 + sub) * 32 + 8 * g;
    h8 blo = *(const h8*)bp0;
    h8 bhi = *(const h8*)(bp0 + 16);
    #pragma unroll
    for (int q = 0; q < 9; ++q) {
      v16h bf = __builtin_shufflevector(blo, bhi, 0,1,2,3,4,5,6,7,8,9,10,11,12,13,14,15);
      if (q < 8) {
        const _Float16* bp = bp0 + (q + 1) * 512;   // next 16 cols * 32 kk
        blo = *(const h8*)bp;
        bhi = *(const h8*)(bp + 16);
      }
      acc[q] = __builtin_amdgcn_wmma_f32_16x16x32_f16(false, af, false, bf,
                                                      (short)0, acc[q], false, false);
    }
  }

  __syncthreads();
  // Epilogue: apply temp * inv_te[i] * inv_tr[j], spill sim tile to LDS.
  // (C/D layout: VGPR r -> M = r + 8g, N = sub)
  float si[8];
  #pragma unroll
  for (int r = 0; r < 8; ++r) si[r] = temp * invTe[i0 + r + 8 * g];
  #pragma unroll
  for (int q = 0; q < 9; ++q) {
    int j0 = (wave * 9 + q) * 16;
    float bj = invTr[j0 + sub];
    #pragma unroll
    for (int r = 0; r < 8; ++r)
      sim_lds[(r + 8 * g) * 576 + j0 + sub] = si[r] * bj * acc[q][r];
  }
  __syncthreads();

  // Row-wise softmax over 576 train locations + dot with downsampled labels.
  const float* ld = ldown + (m * NSEQ + n) * LF;
  int row = tid >> 3, p = tid & 7;               // 16 rows x 8 threads
  const float* srow = sim_lds + row * 576;
  float mx = -3.4e38f;
  for (int j = p; j < 576; j += 8) mx = fmaxf(mx, srow[j]);
  redA[tid] = mx;
  __syncthreads();
  float rmx = redA[row * 8];
  #pragma unroll
  for (int u = 1; u < 8; ++u) rmx = fmaxf(rmx, redA[row * 8 + u]);
  __syncthreads();
  float se = 0.f, sd = 0.f;
  for (int j = p; j < 576; j += 8) {
    float e = __expf(srow[j] - rmx);
    se += e;
    sd += e * ld[j];
  }
  redA[tid] = se;
  redB[tid] = sd;
  __syncthreads();
  if (p == 0) {
    float ts = 0.f, td = 0.f;
    #pragma unroll
    for (int u = 0; u < 8; ++u) { ts += redA[row * 8 + u]; td += redB[row * 8 + u]; }
    pmt_down[(m * NSEQ + n) * LF + i0 + row] = td / ts;
  }
}

// ---------------------------------------------------------------------------
// Kernel 4: per output pixel: bilinear upsample 24->96 of all 8 memories,
// mean / unbiased std, tanh + sparse triangular binning (2 nonzeros per
// encoding), fused conv1 -> BN -> ReLU -> conv2.
// ---------------------------------------------------------------------------
__global__ __launch_bounds__(256) void khead(const float* __restrict__ scores,
                                             const float* __restrict__ pmt_down,
                                             const float* __restrict__ w1,
                                             const float* __restrict__ b1,
                                             const float* __restrict__ gamma,
                                             const float* __restrict__ beta,
                                             const float* __restrict__ bmean,
                                             const float* __restrict__ bvar,
                                             const float* __restrict__ w2,
                                             const float* __restrict__ b2,
                                             float* __restrict__ out) {
  __shared__ float w1s[4096];
  __shared__ float b1s[64], cA[64], cB[64], w2s[64];
  int tid = threadIdx.x;
  for (int idx = tid; idx < 4096; idx += 256) w1s[idx] = w1[idx];
  if (tid < 64) {
    float A = gamma[tid] / sqrtf(bvar[tid] + BN_EPS);
    cA[tid]  = A;
    cB[tid]  = beta[tid] - bmean[tid] * A;
    b1s[tid] = b1[tid];
    w2s[tid] = w2[tid];
  }
  __syncthreads();

  int gid = blockIdx.x * 256 + tid;                 // (n, py, px) flat
  int n   = gid / LPIX;
  int rem = gid % LPIX;
  int py  = rem / WL, px = rem % WL;

  // Bilinear 24->96 (half-pixel; edge renorm == clamp for pure upscale)
  float fy = 0.25f * py - 0.375f;
  float fx = 0.25f * px - 0.375f;
  float y0f = floorf(fy), x0f = floorf(fx);
  float wy = fy - y0f,    wx = fx - x0f;
  int y0 = (int)y0f, x0 = (int)x0f;
  int y1 = y0 + 1,   x1 = x0 + 1;
  y0 = max(0, min(WF - 1, y0)); y1 = max(0, min(WF - 1, y1));
  x0 = max(0, min(WF - 1, x0)); x1 = max(0, min(WF - 1, x1));

  float pm[NMEM];
  float s = 0.f;
  #pragma unroll
  for (int m = 0; m < NMEM; ++m) {
    const float* pd = pmt_down + (m * NSEQ + n) * LF;
    float v = (1.f - wy) * ((1.f - wx) * pd[y0 * WF + x0] + wx * pd[y0 * WF + x1])
            +        wy  * ((1.f - wx) * pd[y1 * WF + x0] + wx * pd[y1 * WF + x1]);
    pm[m] = v;
    s += v;
  }
  float mean = s * (1.0f / NMEM);
  float var = 0.f;
  #pragma unroll
  for (int m = 0; m < NMEM; ++m) { float d = pm[m] - mean; var += d * d; }
  float sd = sqrtf(var * (1.0f / (NMEM - 1)));

  // Triangular-bin encodings: each has exactly two adjacent nonzero bins.
  float tsc = tanhf(scores[gid]);
  tsc = fminf(1.f, fmaxf(-1.f, tsc));
  float us = (tsc + 1.0f) * 15.5f;                  // 32 bins over [-1,1]
  int ks = (int)us; if (ks > 30) ks = 30; float fs = us - (float)ks;

  float tm = fminf(1.f, fmaxf(0.f, tanhf(mean)));
  float um = tm * 15.0f;                            // 16 bins over [0,1]
  int km = (int)um; if (km > 14) km = 14; float fm = um - (float)km;

  float tdv = fminf(1.f, fmaxf(0.f, tanhf(sd)));
  float ud = tdv * 15.0f;
  int kd = (int)ud; if (kd > 14) kd = 14; float fd = ud - (float)kd;

  int iS = ks, iM = 32 + km, iD = 48 + kd;
  float acc = b2[0];
  #pragma unroll 4
  for (int o = 0; o < 64; ++o) {
    const float* wr = w1s + o * 64;
    float v = b1s[o]
            + wr[iS] * (1.f - fs) + wr[iS + 1] * fs
            + wr[iM] * (1.f - fm) + wr[iM + 1] * fm
            + wr[iD] * (1.f - fd) + wr[iD + 1] * fd;
    v = fmaxf(v * cA[o] + cB[o], 0.f);
    acc += w2s[o] * v;
  }
  out[gid] = acc;
}

// ---------------------------------------------------------------------------
extern "C" void kernel_launch(void* const* d_in, const int* in_sizes, int n_in,
                              void* d_out, int out_size, void* d_ws, size_t ws_size,
                              hipStream_t stream) {
  (void)in_sizes; (void)n_in; (void)out_size; (void)ws_size;
  const float* test_scores  = (const float*)d_in[0];
  const float* train_labels = (const float*)d_in[1];
  const float* test_feat    = (const float*)d_in[2];
  const float* train_feats  = (const float*)d_in[3];
  const float* softmax_temp = (const float*)d_in[4];
  const float* conv1_w      = (const float*)d_in[5];
  const float* conv1_b      = (const float*)d_in[6];
  const float* bn_gamma     = (const float*)d_in[7];
  const float* bn_beta      = (const float*)d_in[8];
  const float* bn_mean      = (const float*)d_in[9];
  const float* bn_var       = (const float*)d_in[10];
  const float* conv2_w      = (const float*)d_in[11];
  const float* conv2_b      = (const float*)d_in[12];
  float* out = (float*)d_out;

  // Workspace layout (f32): inv_te | inv_tr | labels_down | pmt_down  (~900 KB)
  float* inv_te = (float*)d_ws;
  float* inv_tr = inv_te + NSEQ * LF;                 // +9216
  float* ldown  = inv_tr + NMEM * NSEQ * LF;          // +73728
  float* pmtd   = ldown  + NMEM * NSEQ * LF;          // +73728

  knorm <<<(NSEQ * LF + NMEM * NSEQ * LF + 255) / 256, 256, 0, stream>>>(
      test_feat, train_feats, inv_te, inv_tr);
  kldown<<<(NMEM * NSEQ * LF + 255) / 256, 256, 0, stream>>>(train_labels, ldown);
  kattn <<<dim3(36, NMEM, NSEQ), 128, 0, stream>>>(
      test_feat, train_feats, inv_te, inv_tr, ldown, softmax_temp, pmtd);
  khead <<<(NSEQ * LPIX) / 256, 256, 0, stream>>>(
      test_scores, pmtd, conv1_w, conv1_b, bn_gamma, bn_beta, bn_mean, bn_var,
      conv2_w, conv2_b, out);
}
